// AlexNetMoELossFree_55095840473660
// MI455X (gfx1250) — matrix operand order repaired and checked by reading, not compiled, not run on hardware
//
#include <hip/hip_runtime.h>
#include <hip/hip_bf16.h>

// ---------------------------------------------------------------------------
// CDNA5 WMMA (wave32): D(16x16 f32) = A(16x32 bf16) x B(32x16 bf16) + C
// ---------------------------------------------------------------------------
typedef __attribute__((ext_vector_type(16))) __bf16 v16bf;
typedef __attribute__((ext_vector_type(8)))  __bf16 v8bf;
typedef __attribute__((ext_vector_type(8)))  float  v8f;

#define LDA 40   // 32 bf16 + 8 pad (80B row stride, 16B-aligned sublocations)
#define LDB 40

// Async global->LDS (gfx1250 GLOBAL_LOAD_ASYNC_TO_LDS_B128, ASYNCcnt-tracked).
// Probe (round-3 stderr) showed the builtin exists with int4-vector-pointer
// params: (addrspace(1) v4i* src, addrspace(3) v4i* dst, imm offset, imm cpol).
#if __has_builtin(__builtin_amdgcn_global_load_async_to_lds_b128) && \
    __has_builtin(__builtin_amdgcn_s_wait_asynccnt)
#define USE_ASYNC_LDS 1
typedef int v4i __attribute__((vector_size(16)));
typedef v4i __attribute__((address_space(1)))* gptr_v4i;
typedef v4i __attribute__((address_space(3)))* lptr_v4i;
#else
#define USE_ASYNC_LDS 0
#endif

__device__ __forceinline__ __bf16 f2bf(float f) { return (__bf16)f; }

__device__ __forceinline__ v16bf cat16(v8bf lo, v8bf hi) {
  return __builtin_shufflevector(lo, hi, 0,1,2,3,4,5,6,7,8,9,10,11,12,13,14,15);
}

// Stage one 16B chunk of packed bf16 weights into LDS.
__device__ __forceinline__ void stage_a16(const __bf16* gsrc, __bf16* ldst) {
#if USE_ASYNC_LDS
  __builtin_amdgcn_global_load_async_to_lds_b128(
      (gptr_v4i)gsrc, (lptr_v4i)ldst, 0, 0);
#else
  *(v8bf*)ldst = *(const v8bf*)gsrc;
#endif
}

__device__ __forceinline__ void wait_async_stage() {
#if USE_ASYNC_LDS
  __builtin_amdgcn_s_wait_asynccnt(0);
#endif
}

// ---------------------------------------------------------------------------
// Weight pre-pack: fp32 [M][K] -> bf16 [Mpad][KP], zero-padded. Branchless
// (clamped addresses + select), 8 elements / thread, vector 16B stores.
// ---------------------------------------------------------------------------
__global__ void __launch_bounds__(256)
pack_w_bf16(const float* __restrict__ in, __bf16* __restrict__ out,
            int M, int K, int Mpad, int KP)
{
  int idx = (blockIdx.x * blockDim.x + threadIdx.x) * 8;
  if (idx >= Mpad * KP) return;
  int row = idx / KP;
  int col = idx - row * KP;
  int rc = row < M ? row : M - 1;
  const float* src = in + (size_t)rc * K;
  v8bf t;
  #pragma unroll
  for (int i = 0; i < 8; ++i) {
    int c  = col + i;
    int cc = c < K ? c : K - 1;
    float v = src[cc];
    t[i] = f2bf((row < M && c < K) ? v : 0.0f);
  }
  *(v8bf*)(out + idx) = t;
}

// ---------------------------------------------------------------------------
// Implicit-GEMM conv + bias + eval-BN + ReLU.  A = packed bf16 weights
// [Mpad][KP]; B = branchless im2col gather (clamped loads + select).
// Block 256 thr = 8 waves (2M x 4N); tile 64x128, BK=32; wave does 2x2 WMMA.
// ---------------------------------------------------------------------------
template<int KS>
__global__ void __launch_bounds__(256)
conv_bn_relu_wmma(const float* __restrict__ x, const __bf16* __restrict__ wpk,
                  const float* __restrict__ cbias, const float* __restrict__ bng,
                  const float* __restrict__ bnb, const float* __restrict__ bnm,
                  const float* __restrict__ bnv, float* __restrict__ y,
                  int Cin, int H, int W, int Cout, int stride, int pad,
                  int OH, int OW, int K, int KP)
{
  __shared__ __align__(16) __bf16 Al[64 * LDA];
  __shared__ __align__(16) __bf16 Bl[128 * LDB];

  const int tid   = threadIdx.x;
  const int lane  = tid & 31;
  const int wave  = tid >> 5;
  const int waveM = wave & 1;
  const int waveN = wave >> 1;
  const int bM  = blockIdx.y * 64;
  const int bN  = blockIdx.x * 128;
  const int OHW = OH * OW;
  const int hw  = H * W;

  // B (im2col) mapping: 128 n-rows x 2 k-halves of 16
  const int nrow = tid >> 1;
  const int ng   = bN + nrow;
  const int bimg = ng / OHW;
  const int prem = ng - bimg * OHW;
  const int oh   = prem / OW;
  const int ow   = prem - oh * OW;
  const int ih0  = oh * stride - pad;
  const int iw0  = ow * stride - pad;
  const int kh16 = (tid & 1) * 16;
  const float* xb = x + (size_t)bimg * Cin * hw;

  // A mapping: 64 m-rows x 4 k-segments of 8 (pure 16B copies, no guards)
  const int arow = tid >> 2;
  const int aseg = (tid & 3) * 8;
  const __bf16* ag = wpk + (size_t)(bM + arow) * KP + aseg;
  __bf16* al = &Al[arow * LDA + aseg];

  v8f acc[2][2] = {};

  for (int k0 = 0; k0 < KP; k0 += 32) {
    // ---- stage A tile: 16B global -> LDS (async if available) ----
    stage_a16(ag + k0, al);
    __builtin_prefetch(ag + k0 + 32, 0, 0);
    // ---- stage B tile: branchless clamped im2col gather ----
    {
      v8bf t0, t1;
      #pragma unroll
      for (int i = 0; i < 16; ++i) {
        int k  = k0 + kh16 + i;
        int kc = (k < K) ? k : 0;
        int ci = kc / (KS * KS);
        int r  = kc - ci * (KS * KS);
        int kh = r / KS;
        int kw = r - kh * KS;
        int ih = ih0 + kh;
        int iw = iw0 + kw;
        bool ok = (k < K) & ((unsigned)ih < (unsigned)H) & ((unsigned)iw < (unsigned)W);
        int ihc = ih < 0 ? 0 : (ih >= H ? H - 1 : ih);
        int iwc = iw < 0 ? 0 : (iw >= W ? W - 1 : iw);
        float v = xb[ci * hw + ihc * W + iwc];   // always in-bounds
        v = ok ? v : 0.0f;
        if (i < 8) t0[i] = f2bf(v); else t1[i - 8] = f2bf(v);
      }
      *(v8bf*)&Bl[nrow * LDB + kh16]     = t0;
      *(v8bf*)&Bl[nrow * LDB + kh16 + 8] = t1;
    }
    wait_async_stage();
    __syncthreads();
    // ---- WMMA compute ----
    {
      const int half = lane >> 4;
      const int r0 = waveM * 32 + (lane & 15);
      const int c0 = waveN * 32 + (lane & 15);
      v16bf afrag[2], bfrag[2];
      #pragma unroll
      for (int t = 0; t < 2; ++t) {
        int row = r0 + t * 16;
        afrag[t] = cat16(*(const v8bf*)&Al[row * LDA + half * 16],
                         *(const v8bf*)&Al[row * LDA + half * 16 + 8]);
        int col = c0 + t * 16;
        bfrag[t] = cat16(*(const v8bf*)&Bl[col * LDB + half * 16],
                         *(const v8bf*)&Bl[col * LDB + half * 16 + 8]);
      }
      #pragma unroll
      for (int tm = 0; tm < 2; ++tm)
        #pragma unroll
        for (int tn = 0; tn < 2; ++tn)
          acc[tm][tn] = __builtin_amdgcn_wmma_f32_16x16x32_bf16(
              false, afrag[tm], false, bfrag[tn], (short)0, acc[tm][tn],
              false, false);
    }
    __syncthreads();
  }

  // ---- epilogue: (acc + conv_b) eval-BN + ReLU, scatter to NCHW ----
  const int half = lane >> 4;
  #pragma unroll
  for (int tm = 0; tm < 2; ++tm) {
    int mbase = bM + waveM * 32 + tm * 16 + half * 8;
    #pragma unroll
    for (int tn = 0; tn < 2; ++tn) {
      int n   = bN + waveN * 32 + tn * 16 + (lane & 15);
      int bi  = n / OHW;
      int r   = n - bi * OHW;
      int oh2 = r / OW;
      int ow2 = r - oh2 * OW;
      #pragma unroll
      for (int v = 0; v < 8; ++v) {
        int m = mbase + v;
        if (m < Cout) {
          float s    = bng[m] * rsqrtf(bnv[m] + 1e-5f);
          float bias = (cbias[m] - bnm[m]) * s + bnb[m];
          float val  = fmaxf(acc[tm][tn][v] * s + bias, 0.0f);
          y[(((size_t)bi * Cout + m) * OH + oh2) * OW + ow2] = val;
        }
      }
    }
  }
}

// ---------------------------------------------------------------------------
// FC GEMM: Y[n][m] = act( X[n][:] . W[m][:] + bias[m] )
//   A = packed bf16 weights [M][K]; B = fp32 activations [N][K].
//   Requires M%64==0, N%128==0, K%32==0 (holds: 4096x{256}x{9216,4096}).
// ---------------------------------------------------------------------------
__global__ void __launch_bounds__(256)
fc_wmma(const float* __restrict__ X, const __bf16* __restrict__ wpk,
        const float* __restrict__ bias, float* __restrict__ Y,
        int M, int N, int K, int relu)
{
  __shared__ __align__(16) __bf16 Al[64 * LDA];
  __shared__ __align__(16) __bf16 Bl[128 * LDB];

  const int tid   = threadIdx.x;
  const int lane  = tid & 31;
  const int wave  = tid >> 5;
  const int waveM = wave & 1;
  const int waveN = wave >> 1;
  const int bM = blockIdx.y * 64;
  const int bN = blockIdx.x * 128;

  const int nrow = tid >> 1;
  const int kh16 = (tid & 1) * 16;
  const float* xg = X + (size_t)(bN + nrow) * K + kh16;

  const int arow = tid >> 2;
  const int aseg = (tid & 3) * 8;
  const __bf16* ag = wpk + (size_t)(bM + arow) * K + aseg;
  __bf16* al = &Al[arow * LDA + aseg];

  v8f acc[2][2] = {};

  for (int k0 = 0; k0 < K; k0 += 32) {
    stage_a16(ag + k0, al);
    __builtin_prefetch(ag + k0 + 32, 0, 0);
    {
      v8bf t0, t1;
      #pragma unroll
      for (int q = 0; q < 4; ++q) {
        float4 f = *(const float4*)(xg + k0 + q * 4);
        if (q < 2) {
          t0[q * 4 + 0] = f2bf(f.x); t0[q * 4 + 1] = f2bf(f.y);
          t0[q * 4 + 2] = f2bf(f.z); t0[q * 4 + 3] = f2bf(f.w);
        } else {
          t1[(q - 2) * 4 + 0] = f2bf(f.x); t1[(q - 2) * 4 + 1] = f2bf(f.y);
          t1[(q - 2) * 4 + 2] = f2bf(f.z); t1[(q - 2) * 4 + 3] = f2bf(f.w);
        }
      }
      *(v8bf*)&Bl[nrow * LDB + kh16]     = t0;
      *(v8bf*)&Bl[nrow * LDB + kh16 + 8] = t1;
    }
    wait_async_stage();
    __syncthreads();
    {
      const int half = lane >> 4;
      const int r0 = waveM * 32 + (lane & 15);
      const int c0 = waveN * 32 + (lane & 15);
      v16bf afrag[2], bfrag[2];
      #pragma unroll
      for (int t = 0; t < 2; ++t) {
        int row = r0 + t * 16;
        afrag[t] = cat16(*(const v8bf*)&Al[row * LDA + half * 16],
                         *(const v8bf*)&Al[row * LDA + half * 16 + 8]);
        int col = c0 + t * 16;
        bfrag[t] = cat16(*(const v8bf*)&Bl[col * LDB + half * 16],
                         *(const v8bf*)&Bl[col * LDB + half * 16 + 8]);
      }
      #pragma unroll
      for (int tm = 0; tm < 2; ++tm)
        #pragma unroll
        for (int tn = 0; tn < 2; ++tn)
          acc[tm][tn] = __builtin_amdgcn_wmma_f32_16x16x32_bf16(
              false, afrag[tm], false, bfrag[tn], (short)0, acc[tm][tn],
              false, false);
    }
    __syncthreads();
  }

  const int half = lane >> 4;
  #pragma unroll
  for (int tm = 0; tm < 2; ++tm) {
    int mbase = bM + waveM * 32 + tm * 16 + half * 8;
    #pragma unroll
    for (int tn = 0; tn < 2; ++tn) {
      int n = bN + waveN * 32 + tn * 16 + (lane & 15);
      #pragma unroll
      for (int v = 0; v < 8; ++v) {
        int m = mbase + v;
        float val = acc[tm][tn][v] + bias[m];
        if (relu) val = fmaxf(val, 0.0f);
        Y[(size_t)n * M + m] = val;
      }
    }
  }
}

// ---------------------------------------------------------------------------
// 3x3 stride-2 max pool (VALID; windows always in-bounds for these shapes)
// ---------------------------------------------------------------------------
__global__ void maxpool3x3s2(const float* __restrict__ in, float* __restrict__ out,
                             int C, int H, int W, int OH, int OW, int total)
{
  int idx = blockIdx.x * blockDim.x + threadIdx.x;
  if (idx >= total) return;
  int ow = idx % OW;
  int t  = idx / OW;
  int oh = t % OH;
  t /= OH;
  int c = t % C;
  int b = t / C;
  const float* p = in + (((size_t)b * C + c) * H + oh * 2) * W + ow * 2;
  float m = -3.4e38f;
  #pragma unroll
  for (int r = 0; r < 3; ++r)
    #pragma unroll
    for (int s = 0; s < 3; ++s)
      m = fmaxf(m, p[r * W + s]);
  out[idx] = m;
}

// ---------------------------------------------------------------------------
// Gate: scores = h @ gate_w.T + gate_b; chosen = argmax(scores + expert_bias)
// ---------------------------------------------------------------------------
__global__ void gate_argmax(const float* __restrict__ h, const float* __restrict__ gw,
                            const float* __restrict__ gb, const float* __restrict__ ebias,
                            int* __restrict__ chosen)
{
  __shared__ float part[3][32];
  __shared__ float score[3];
  const int b = blockIdx.x, tid = threadIdx.x;
  const int lane = tid & 31, wave = tid >> 5;
  if (wave < 3) {
    const float* hb = h + (size_t)b * 4096;
    const float* wr = gw + (size_t)wave * 4096;
    float s = 0.0f;
    for (int k = lane; k < 4096; k += 32) s += hb[k] * wr[k];
    part[wave][lane] = s;
  }
  __syncthreads();
  if (tid < 3) {
    float s = 0.0f;
    #pragma unroll
    for (int l = 0; l < 32; ++l) s += part[tid][l];
    score[tid] = s + gb[tid] + ebias[tid];
  }
  __syncthreads();
  if (tid == 0) {
    int best = 0; float bs = score[0];
    if (score[1] > bs) { bs = score[1]; best = 1; }
    if (score[2] > bs) { bs = score[2]; best = 2; }
    chosen[b] = best;
  }
}

// ---------------------------------------------------------------------------
// Expert head: out[b][c] = h[b][:] . exp_w[e][:, c] + exp_b[e][c]
// ---------------------------------------------------------------------------
__global__ void expert_head(const float* __restrict__ h, const float* __restrict__ ew,
                            const float* __restrict__ eb, const int* __restrict__ chosen,
                            float* __restrict__ out)
{
  const int b = blockIdx.x, tid = threadIdx.x;
  if (tid >= 100) return;
  const int e = chosen[b];
  const float* hb = h + (size_t)b * 4096;
  const float* wp = ew + (size_t)e * 4096 * 100 + tid;
  float acc = eb[e * 100 + tid];
  for (int k = 0; k < 4096; k += 4) {
    acc += hb[k + 0] * wp[(size_t)(k + 0) * 100];
    acc += hb[k + 1] * wp[(size_t)(k + 1) * 100];
    acc += hb[k + 2] * wp[(size_t)(k + 2) * 100];
    acc += hb[k + 3] * wp[(size_t)(k + 3) * 100];
  }
  out[b * 100 + tid] = acc;
}

// ---------------------------------------------------------------------------
// Host-side orchestration
// ---------------------------------------------------------------------------
extern "C" void kernel_launch(void* const* d_in, const int* in_sizes, int n_in,
                              void* d_out, int out_size, void* d_ws, size_t ws_size,
                              hipStream_t stream)
{
  (void)in_sizes; (void)n_in; (void)out_size; (void)ws_size;

  const float* x = (const float*)d_in[0];
  const float* cw[5], *cb[5], *g_[5], *bb_[5], *bm_[5], *bv_[5];
  for (int i = 0; i < 5; ++i) {
    cw[i]  = (const float*)d_in[1 + i];
    cb[i]  = (const float*)d_in[6 + i];
    g_[i]  = (const float*)d_in[11 + i];
    bb_[i] = (const float*)d_in[16 + i];
    bm_[i] = (const float*)d_in[21 + i];
    bv_[i] = (const float*)d_in[26 + i];
  }
  const float* fc1_w = (const float*)d_in[31];
  const float* fc1_b = (const float*)d_in[32];
  const float* fc2_w = (const float*)d_in[33];
  const float* fc2_b = (const float*)d_in[34];
  const float* gate_w = (const float*)d_in[35];
  const float* gate_b = (const float*)d_in[36];
  const float* exp_w = (const float*)d_in[37];
  const float* exp_b = (const float*)d_in[38];
  const float* ebias = (const float*)d_in[39];

  // scratch: A @0 (<=298MB), B @300MiB (<=72MB), packed weights @384MiB (~117MB)
  char* ws = (char*)d_ws;
  float* bufs[2] = { (float*)ws, (float*)(ws + 300ull * 1024 * 1024) };
  __bf16* wbase = (__bf16*)(ws + 384ull * 1024 * 1024);
  int t = 0;

  static const int CFG[5][9] = {
    // Cin, H,   W,   Cout, KS, stride, pad, OH, OW
    {   3, 227, 227,   96, 11, 4, 0, 55, 55 },
    {  96,  27,  27,  256,  5, 1, 2, 27, 27 },
    { 256,  13,  13,  384,  3, 1, 1, 13, 13 },
    { 384,  13,  13,  384,  3, 1, 1, 13, 13 },
    { 384,  13,  13,  256,  3, 1, 1, 13, 13 },
  };
  static const bool POOL[5] = { true, true, false, false, true };
  const int B = 256;

  // ---- pack all weights (5 conv + 2 fc) to zero-padded bf16 ----
  __bf16* wpk[7];
  {
    const float* src[7] = { cw[0], cw[1], cw[2], cw[3], cw[4], fc1_w, fc2_w };
    int Ms[7], Ks[7];
    for (int i = 0; i < 5; ++i) {
      Ms[i] = CFG[i][3];
      Ks[i] = CFG[i][0] * CFG[i][4] * CFG[i][4];
    }
    Ms[5] = 4096; Ks[5] = 9216;
    Ms[6] = 4096; Ks[6] = 4096;
    size_t off = 0;
    for (int i = 0; i < 7; ++i) {
      int Mpad = (Ms[i] + 63) & ~63;
      int KP   = (Ks[i] + 31) & ~31;
      wpk[i] = wbase + off;
      int total8 = (Mpad * KP) / 8;
      pack_w_bf16<<<(total8 + 255) / 256, 256, 0, stream>>>(
          src[i], wpk[i], Ms[i], Ks[i], Mpad, KP);
      off += (size_t)Mpad * KP;
    }
  }

  // ---- conv backbone ----
  const float* cur = x;
  for (int i = 0; i < 5; ++i) {
    int Cin = CFG[i][0], H = CFG[i][1], W = CFG[i][2], Cout = CFG[i][3];
    int KS = CFG[i][4], st = CFG[i][5], pd = CFG[i][6], OH = CFG[i][7], OW = CFG[i][8];
    int K  = Cin * KS * KS;
    int KP = (K + 31) & ~31;
    int Mpad = (Cout + 63) & ~63;
    int N = B * OH * OW;                 // always a multiple of 128
    float* yout = bufs[t]; t ^= 1;
    dim3 grid(N / 128, Mpad / 64);
    if (KS == 11)
      conv_bn_relu_wmma<11><<<grid, 256, 0, stream>>>(
          cur, wpk[i], cb[i], g_[i], bb_[i], bm_[i], bv_[i], yout,
          Cin, H, W, Cout, st, pd, OH, OW, K, KP);
    else if (KS == 5)
      conv_bn_relu_wmma<5><<<grid, 256, 0, stream>>>(
          cur, wpk[i], cb[i], g_[i], bb_[i], bm_[i], bv_[i], yout,
          Cin, H, W, Cout, st, pd, OH, OW, K, KP);
    else
      conv_bn_relu_wmma<3><<<grid, 256, 0, stream>>>(
          cur, wpk[i], cb[i], g_[i], bb_[i], bm_[i], bv_[i], yout,
          Cin, H, W, Cout, st, pd, OH, OW, K, KP);
    cur = yout;
    if (POOL[i]) {
      int POH = (OH - 3) / 2 + 1, POW = (OW - 3) / 2 + 1;
      int total = B * Cout * POH * POW;
      float* pout = bufs[t]; t ^= 1;
      maxpool3x3s2<<<(total + 255) / 256, 256, 0, stream>>>(
          cur, pout, Cout, OH, OW, POH, POW, total);
      cur = pout;
    }
  }

  // ---- FC trunk ----
  float* h1 = bufs[t]; t ^= 1;
  fc_wmma<<<dim3(256 / 128, 4096 / 64), 256, 0, stream>>>(
      cur, wpk[5], fc1_b, h1, 4096, 256, 9216, 1);
  float* h2 = bufs[t]; t ^= 1;
  fc_wmma<<<dim3(256 / 128, 4096 / 64), 256, 0, stream>>>(
      h1, wpk[6], fc2_b, h2, 4096, 256, 4096, 1);

  // ---- MoE head ----
  int* chosen = (int*)bufs[t];
  gate_argmax<<<256, 128, 0, stream>>>(h2, gate_w, gate_b, ebias, chosen);
  expert_head<<<256, 128, 0, stream>>>(h2, exp_w, exp_b, chosen, (float*)d_out);
}
